// DglGraphAttentionNetwork_21715354648949
// MI455X (gfx1250) — compile-verified, matrix operand order checked by reading, not run on hardware
//
#include <hip/hip_runtime.h>
#include <hip/hip_bf16.h>
#include <stdint.h>

// GAT layer for MI455X (gfx1250, wave32, WMMA + TDM).
// h = text @ (weight @ fc_w); then per-head segment softmax + scatter-add.

#define DIN 512
#define NHEAD 4
#define DHEAD 128
#define NEG_SLOPE 0.2f

typedef __attribute__((ext_vector_type(16))) __bf16 v16bf;
typedef __attribute__((ext_vector_type(8)))  float  v8f;
typedef __attribute__((ext_vector_type(4)))  unsigned int v4u;
typedef __attribute__((ext_vector_type(8)))  int v8i_t;
typedef __attribute__((ext_vector_type(4)))  int v4i_t;

union BfFrag { v16bf v; uint4 u[2]; };

#if __has_builtin(__builtin_amdgcn_tensor_load_to_lds) && \
    __has_builtin(__builtin_amdgcn_s_wait_tensorcnt)
#define USE_TDM 1
#else
#define USE_TDM 0
#endif

__device__ __forceinline__ unsigned short f2bf(float x) {
    union { float f; unsigned u; } c; c.f = x;
    unsigned u = c.u;
    unsigned r = u + 0x7FFFu + ((u >> 16) & 1u);   // round-to-nearest-even
    return (unsigned short)(r >> 16);
}

// order-preserving float<->uint encoding for atomicMax on floats
__device__ __forceinline__ unsigned fenc(float f) {
    union { float f; unsigned u; } c; c.f = f;
    return (c.u & 0x80000000u) ? ~c.u : (c.u | 0x80000000u);
}
__device__ __forceinline__ float fdec(unsigned u) {
    union { float f; unsigned u; } c;
    c.u = (u & 0x80000000u) ? (u & 0x7FFFFFFFu) : ~u;
    return c.f;
}

// ---------------------------------------------------------------------------
// K1: Wc = weight @ fc_w, stored as bf16 in WMMA-B swizzled layout.
// For K-tile kt (32 K-rows) and column j, the 32 K-values of that column are
// contiguous in order [k0..7, k16..23, k8..15, k24..31] so a lane reads its 16
// fragment elements as two contiguous uint4 loads, and a block's whole B tile
// (columns n0..n0+63, all kt) is sixteen contiguous 4KB runs (TDM-friendly).
// grid = (512, 2), block = 256
__global__ __launch_bounds__(256)
void k_wcomb(const float* __restrict__ weight, const float* __restrict__ fcw,
             unsigned short* __restrict__ Bswz) {
    int d = blockIdx.x;                                // K row of Wc
    int j = blockIdx.y * 256 + threadIdx.x;            // N col of Wc
    float acc = 0.f;
    const float* wrow = weight + (size_t)d * DIN;
    for (int e = 0; e < DIN; ++e)
        acc = fmaf(wrow[e], fcw[(size_t)e * DIN + j], acc);
    int kt = d >> 5, kr = d & 31;
    int grp = kr >> 3;
    int p = kr + (grp == 1 ? 8 : (grp == 2 ? -8 : 0));
    Bswz[(((size_t)kt * DIN + j) << 5) + p] = f2bf(acc);
}

// K2: float -> bf16, 4 elements / thread
__global__ __launch_bounds__(256)
void k_cvt_bf16(const float* __restrict__ in, unsigned short* __restrict__ out, int n4) {
    int i = blockIdx.x * 256 + threadIdx.x;
    if (i >= n4) return;
    float4 v = ((const float4*)in)[i];
    ushort4 o;
    o.x = f2bf(v.x); o.y = f2bf(v.y); o.z = f2bf(v.z); o.w = f2bf(v.w);
    ((ushort4*)out)[i] = o;
}

// ---------------------------------------------------------------------------
// K3: H = A(bf16)[N,512] @ Bswz(bf16)[512,512] -> f32 [N,512]
// block = 256 (8 waves); wave tile 16(M) x 64(N); block tile 128 x 64.
// B tile (64KB) staged in LDS once per block via the Tensor Data Mover with
// 8B padding per 64B column (72B stride) to spread LDS banks; all 8 waves
// then feed WMMA from ds_load_b128 instead of 8x-redundant global loads.
// All 4 B fragments of a K-step are loaded into distinct registers before the
// WMMA sequence so waits retire against partial DScnt instead of full drains.
// Dynamic LDS: 16*64*72 = 73728 bytes.
// grid = (512/64, N/128)
#define B_COL_STRIDE 72           // 64B data + 8B TDM pad
#define LDS_BYTES (16 * 64 * B_COL_STRIDE)

__global__ __launch_bounds__(256)
void k_gemm_wmma(const unsigned short* __restrict__ A,
                 const unsigned short* __restrict__ Bswz,
                 float* __restrict__ Hout) {
    extern __shared__ __align__(16) char smem[];
    const int lane = threadIdx.x & 31;
    const int wave = threadIdx.x >> 5;
    const int m0   = blockIdx.y * 128 + wave * 16;
    const int n0   = blockIdx.x * 64;
    const int half = lane >> 4;        // K-half select per ISA A/B layout
    const int lm   = lane & 15;

#if USE_TDM
    if (wave == 0) {
        // --- Tensor DMA descriptor (D#), 2D tile, bf16 elements ---
        unsigned long long gaddr =
            (unsigned long long)(uintptr_t)Bswz + (unsigned long long)n0 * 32u * 2u;
        unsigned ldsa = (unsigned)(uintptr_t)smem;
        // group0: count=1 | lds_addr | global_addr | type=2
        v4u g0 = { 1u,
                   ldsa,
                   (unsigned)gaddr,
                   (unsigned)((gaddr >> 32) & 0x01FFFFFFu) | 0x80000000u };
        // group1: data_size=2B, pad_enable, pad_interval=16dw, pad_amount=2dw,
        //         tensor_dim0=2048, tensor_dim1=16, tile_dim0=2048, tile_dim1=16,
        //         tensor_dim0_stride=16384 elements
        v8i_t g1 = { (int)((1u << 16) | (1u << 20) | (3u << 22) | (1u << 25)),
                     (int)(2048u << 16),
                     (int)(16u << 16),
                     (int)(2048u << 16),
                     16,
                     16384,
                     0, 0 };
        v4i_t g2 = { 0, 0, 0, 0 };
        v4i_t g3 = { 0, 0, 0, 0 };
        v8i_t g4 = { 0, 0, 0, 0, 0, 0, 0, 0 };
        __builtin_amdgcn_tensor_load_to_lds(g0, g1, g2, g3, g4, 0);
        __builtin_amdgcn_s_wait_tensorcnt(0);
    }
    __syncthreads();
#endif

    v8f acc[4];
#pragma unroll
    for (int i = 0; i < 4; ++i) acc[i] = (v8f){0.f,0.f,0.f,0.f,0.f,0.f,0.f,0.f};

    const uint4* arow = (const uint4*)(A + (size_t)(m0 + lm) * DIN); // 64 uint4/row

    BfFrag a;
    a.u[0] = arow[half];
    a.u[1] = arow[2 + half];

#pragma unroll 4
    for (int kt = 0; kt < DIN / 32; ++kt) {
        // prefetch next A fragment while WMMAs consume the current one
        int ktn = kt + 1 < DIN / 32 ? kt + 1 : kt;
        BfFrag an;
        an.u[0] = arow[ktn * 4 + half];
        an.u[1] = arow[ktn * 4 + 2 + half];

        // load all four B fragments first (8 ds_load_b128 in flight),
        // then issue the four WMMAs against draining DScnt
        BfFrag b[4];
#pragma unroll
        for (int i = 0; i < 4; ++i) {
#if USE_TDM
            int cc = kt * 64 + i * 16 + lm;                 // column within tile
            const uint4* bp =
                (const uint4*)(smem + cc * B_COL_STRIDE + (half << 5));
#else
            int j = n0 + i * 16 + lm;
            const uint4* bp =
                (const uint4*)(Bswz + ((((size_t)kt * DIN + j) << 5) + (half << 4)));
#endif
            b[i].u[0] = bp[0];
            b[i].u[1] = bp[1];
        }
#pragma unroll
        for (int i = 0; i < 4; ++i)
            acc[i] = __builtin_amdgcn_wmma_f32_16x16x32_bf16(
                false, a.v, false, b[i].v, (short)0, acc[i], false, false);
        a = an;
    }

    // D layout: lane L holds N = L&15; VGPR v holds M = v + (L>>4)*8
#pragma unroll
    for (int i = 0; i < 4; ++i) {
        int j = n0 + i * 16 + lm;
        float* orow = Hout + (size_t)(m0 + half * 8) * DIN + j;
#pragma unroll
        for (int v = 0; v < 8; ++v)
            orow[(size_t)v * DIN] = acc[i][v];
    }
}

// ---------------------------------------------------------------------------
// K4: el[n,h] = <h[n,h,:], attn_l[h,:]>, er likewise. 1 block/node, 1 wave/head.
__global__ __launch_bounds__(128)
void k_elr(const float* __restrict__ Hf, const float* __restrict__ al,
           const float* __restrict__ ar, float* __restrict__ el,
           float* __restrict__ er) {
    int n    = blockIdx.x;
    int head = threadIdx.x >> 5;
    int lane = threadIdx.x & 31;
    float4 hv = ((const float4*)(Hf + (size_t)n * DIN + head * DHEAD))[lane];
    float4 av = ((const float4*)(al + head * DHEAD))[lane];
    float4 bv = ((const float4*)(ar + head * DHEAD))[lane];
    float sl = hv.x * av.x + hv.y * av.y + hv.z * av.z + hv.w * av.w;
    float sr = hv.x * bv.x + hv.y * bv.y + hv.z * bv.z + hv.w * bv.w;
#pragma unroll
    for (int m = 16; m >= 1; m >>= 1) {
        sl += __shfl_xor(sl, m, 32);
        sr += __shfl_xor(sr, m, 32);
    }
    if (lane == 0) {
        el[n * NHEAD + head] = sl;
        er[n * NHEAD + head] = sr;
    }
}

// K5: out = bias broadcast; zero softmax max/denom buffers.
__global__ __launch_bounds__(256)
void k_init(float* __restrict__ out, const float* __restrict__ bias,
            unsigned* __restrict__ maxenc, float* __restrict__ denom, int ntot) {
    int i = blockIdx.x * 256 + threadIdx.x;
    if (i >= ntot) return;
    int c = i & (DIN - 1);
    out[i] = bias[c];
    if (c < NHEAD) {
        int n = i >> 9;
        maxenc[n * NHEAD + c] = 0u;   // == encoded "very negative"
        denom[n * NHEAD + c]  = 0.f;
    }
}

// K6: per (edge,head): e = leaky_relu(el[src]+er[dst]); segment max via atomicMax
__global__ __launch_bounds__(256)
void k_edge_max(const int* __restrict__ src, const int* __restrict__ dst,
                const float* __restrict__ el, const float* __restrict__ er,
                float* __restrict__ elog, unsigned* __restrict__ maxenc, int e4) {
    int i = blockIdx.x * 256 + threadIdx.x;
    if (i >= e4) return;
    int e = i >> 2, hh = i & 3;
    int s = src[e], d0 = dst[e];
    float v = el[s * NHEAD + hh] + er[d0 * NHEAD + hh];
    v = v > 0.f ? v : NEG_SLOPE * v;
    elog[i] = v;
    atomicMax(&maxenc[d0 * NHEAD + hh], fenc(v));
}

// K7: ex = exp(e - max); denom += ex
__global__ __launch_bounds__(256)
void k_edge_exp(const int* __restrict__ dst, const unsigned* __restrict__ maxenc,
                float* __restrict__ elog, float* __restrict__ denom, int e4) {
    int i = blockIdx.x * 256 + threadIdx.x;
    if (i >= e4) return;
    int e = i >> 2, hh = i & 3;
    int d0 = dst[e];
    float ex = __expf(elog[i] - fdec(maxenc[d0 * NHEAD + hh]));
    elog[i] = ex;
    unsafeAtomicAdd(&denom[d0 * NHEAD + hh], ex);
}

// K8: out[dst,h,:] += alpha * h[src,h,:]; one wave per edge, float4 per lane.
// h and out are L2-resident (33.5MB each vs 192MB L2) -> atomics hit L2.
__global__ __launch_bounds__(256)
void k_agg(const int* __restrict__ src, const int* __restrict__ dst,
           const float* __restrict__ Hf, const float* __restrict__ elog,
           const float* __restrict__ denom, float* __restrict__ out, int E) {
    int e = blockIdx.x * 8 + (threadIdx.x >> 5);
    if (e >= E) return;
    int lane = threadIdx.x & 31;
    int s = src[e], d0 = dst[e];
#pragma unroll
    for (int hh = 0; hh < NHEAD; ++hh) {
        float alpha = elog[e * NHEAD + hh] / denom[d0 * NHEAD + hh];
        float4 hv = ((const float4*)(Hf + (size_t)s * DIN + hh * DHEAD))[lane];
        float* op = out + (size_t)d0 * DIN + hh * DHEAD + lane * 4;
        unsafeAtomicAdd(op + 0, alpha * hv.x);
        unsafeAtomicAdd(op + 1, alpha * hv.y);
        unsafeAtomicAdd(op + 2, alpha * hv.z);
        unsafeAtomicAdd(op + 3, alpha * hv.w);
    }
}

// ---------------------------------------------------------------------------
extern "C" void kernel_launch(void* const* d_in, const int* in_sizes, int n_in,
                              void* d_out, int out_size, void* d_ws, size_t ws_size,
                              hipStream_t stream) {
    const float* text   = (const float*)d_in[0];
    const float* weight = (const float*)d_in[1];
    const float* fcw    = (const float*)d_in[2];
    const float* attn_l = (const float*)d_in[3];
    const float* attn_r = (const float*)d_in[4];
    const float* bias   = (const float*)d_in[5];
    const int*   src    = (const int*)d_in[6];
    const int*   dst    = (const int*)d_in[7];
    float*       out    = (float*)d_out;

    const int N = in_sizes[0] / DIN;   // 16384
    const int E = in_sizes[6];         // 278528

    char* ws = (char*)d_ws;
    size_t off = 0;
    auto alloc = [&](size_t bytes) {
        off = (off + 255) & ~(size_t)255;
        size_t p = off; off += bytes; return p;
    };
    unsigned short* Bswz  = (unsigned short*)(ws + alloc((size_t)DIN * DIN * 2));
    unsigned short* Abf   = (unsigned short*)(ws + alloc((size_t)N * DIN * 2));
    float*          Hf    = (float*)(ws + alloc((size_t)N * DIN * 4));
    float*          el    = (float*)(ws + alloc((size_t)N * NHEAD * 4));
    float*          er    = (float*)(ws + alloc((size_t)N * NHEAD * 4));
    unsigned*       maxen = (unsigned*)(ws + alloc((size_t)N * NHEAD * 4));
    float*          denom = (float*)(ws + alloc((size_t)N * NHEAD * 4));
    float*          elog  = (float*)(ws + alloc((size_t)E * NHEAD * 4));
    (void)ws_size; (void)n_in; (void)out_size;

    // 1) Wc = weight @ fc_w  (bf16, WMMA/TDM-swizzled)
    k_wcomb<<<dim3(DIN, 2), 256, 0, stream>>>(weight, fcw, Bswz);
    // 2) text -> bf16
    int n4 = N * DIN / 4;
    k_cvt_bf16<<<(n4 + 255) / 256, 256, 0, stream>>>(text, Abf, n4);
    // 3) h = text @ Wc   (bf16 WMMA, f32 accumulate; B tile via TDM->LDS)
    k_gemm_wmma<<<dim3(DIN / 64, N / 128), 256, LDS_BYTES, stream>>>(Abf, Bswz, Hf);
    // 4) el / er
    k_elr<<<N, 128, 0, stream>>>(Hf, attn_l, attn_r, el, er);
    // 5) init out with bias; zero max/denom
    int ntot = N * DIN;
    k_init<<<(ntot + 255) / 256, 256, 0, stream>>>(out, bias, maxen, denom, ntot);
    // 6) edge logits + segment max
    int e4 = E * NHEAD;
    k_edge_max<<<(e4 + 255) / 256, 256, 0, stream>>>(src, dst, el, er, elog, maxen, e4);
    // 7) exp + denom
    k_edge_exp<<<(e4 + 255) / 256, 256, 0, stream>>>(dst, maxen, elog, denom, e4);
    // 8) weighted scatter-add (bias already in out)
    k_agg<<<(E + 7) / 8, 256, 0, stream>>>(src, dst, Hf, elog, denom, out, E);
}
